// LlamaAttn_6777458393992
// MI455X (gfx1250) — compile-verified, matrix operand order checked by reading, not run on hardware
//
#include <hip/hip_runtime.h>

// ---------------------------------------------------------------------------
// MI455X (gfx1250) Llama3 attention layer, compile-only target.
// All matmuls run on v_wmma_f32_16x16x32_bf16 (the fast CDNA5 matrix path);
// fp32 inputs are cast to bf16 once (bandwidth-trivial at 23.3 TB/s).
// ---------------------------------------------------------------------------

typedef __attribute__((ext_vector_type(16))) __bf16 bf16x16;
typedef __attribute__((ext_vector_type(8)))  __bf16 bf16x8;
typedef __attribute__((ext_vector_type(8)))  float  f32x8;

#define BATCH      2
#define SEQ        2048
#define MROWS      (BATCH * SEQ)      // 4096
#define HDIM       4096
#define QKVD       6144               // 32*128 + 2*8*128
#define HEAD_DIM   128
#define NQH        32
#define NKV        8
#define QOFF       0
#define KOFF       4096
#define VOFF       5120

__device__ __forceinline__ bf16x16 make_frag(const __bf16* p_lo, const __bf16* p_hi) {
    bf16x16 f;
    *(bf16x8*)&f       = *(const bf16x8*)p_lo;   // VGPR 0-3 : K = half*8 .. +7
    *((bf16x8*)&f + 1) = *(const bf16x8*)p_hi;   // VGPR 4-7 : K = 16+half*8 .. +7
    return f;
}

// ---------------------------------------------------------------------------
// fp32 -> bf16 cast (grid-stride)
// ---------------------------------------------------------------------------
__global__ void cast_bf16(const float* __restrict__ s, __bf16* __restrict__ d, long n) {
    long stride = (long)gridDim.x * blockDim.x;
    for (long i = (long)blockIdx.x * blockDim.x + threadIdx.x; i < n; i += stride)
        d[i] = (__bf16)s[i];
}

// fp32 [R][C] -> bf16 [C][R]  (pre-transpose weights so GEMM B-staging is
// contiguous along K, same as A: every WMMA fragment = two ds_load_b128)
__global__ void transpose_bf16(const float* __restrict__ s, __bf16* __restrict__ d,
                               int R, int C) {
    long n = (long)R * C;
    long stride = (long)gridDim.x * blockDim.x;
    for (long i = (long)blockIdx.x * blockDim.x + threadIdx.x; i < n; i += stride) {
        long r = i / C, c = i % C;
        d[c * (long)R + r] = (__bf16)s[i];
    }
}

// ---------------------------------------------------------------------------
// Tiled bf16 WMMA GEMM:  C[M][N] = A[M][K] * BT[N][K]^T
// Block tile 128x128x32, 256 threads = 8 waves in a 2(M) x 4(N) grid,
// each wave owns 4x2 = 8 accumulator tiles of 16x16.
// ---------------------------------------------------------------------------
template <bool OUT_BF16>
__global__ void __launch_bounds__(256)
gemm_bf16_wmma(const __bf16* __restrict__ A, const __bf16* __restrict__ BT,
               void* __restrict__ Cout, int M, int N, int K) {
    __shared__ __attribute__((aligned(16))) __bf16 As[128][40];  // +8 pad, 16B-aligned rows
    __shared__ __attribute__((aligned(16))) __bf16 Bs[128][40];

    const int tid  = threadIdx.x;
    const int lane = tid & 31;
    const int wave = tid >> 5;
    const int wm   = wave >> 2;            // 0..1
    const int wn   = wave & 3;             // 0..3
    const int half = lane >> 4;            // upper/lower 16 lanes (wave32)
    const int ln   = lane & 15;

    const int row0 = blockIdx.y * 128;
    const int col0 = blockIdx.x * 128;

    const int sr = tid >> 1;               // 0..127 : staging row
    const int sk = (tid & 1) * 16;         // 0 / 16 : staging k-chunk

    f32x8 acc[4][2] = {};

    for (int k0 = 0; k0 < K; k0 += 32) {
        // --- stage A + BT tiles (both row-major along K) -------------------
        const __bf16* asrc = A  + (size_t)(row0 + sr) * K + k0 + sk;
        const __bf16* bsrc = BT + (size_t)(col0 + sr) * K + k0 + sk;
        *(bf16x8*)&As[sr][sk]     = *(const bf16x8*)(asrc);
        *(bf16x8*)&As[sr][sk + 8] = *(const bf16x8*)(asrc + 8);
        *(bf16x8*)&Bs[sr][sk]     = *(const bf16x8*)(bsrc);
        *(bf16x8*)&Bs[sr][sk + 8] = *(const bf16x8*)(bsrc + 8);
        if (k0 + 32 < K) {                 // gfx1250 global_prefetch_b8
            __builtin_prefetch(asrc + 32, 0, 3);
            __builtin_prefetch(bsrc + 32, 0, 3);
        }
        __syncthreads();

        // --- fragments + 8 WMMAs per wave ----------------------------------
        bf16x16 af[4], bf[2];
#pragma unroll
        for (int mi = 0; mi < 4; ++mi) {
            const int r = wm * 64 + mi * 16 + ln;
            af[mi] = make_frag(&As[r][half * 8], &As[r][16 + half * 8]);
        }
#pragma unroll
        for (int ni = 0; ni < 2; ++ni) {
            const int r = wn * 32 + ni * 16 + ln;
            bf[ni] = make_frag(&Bs[r][half * 8], &Bs[r][16 + half * 8]);
        }
#pragma unroll
        for (int mi = 0; mi < 4; ++mi)
#pragma unroll
            for (int ni = 0; ni < 2; ++ni)
                acc[mi][ni] = __builtin_amdgcn_wmma_f32_16x16x32_bf16(
                    false, af[mi], false, bf[ni], (short)0, acc[mi][ni], false, false);
        __syncthreads();
    }

    // --- writeback: C layout = lane -> col, VGPR r -> row (+8 for upper half)
#pragma unroll
    for (int mi = 0; mi < 4; ++mi)
#pragma unroll
        for (int ni = 0; ni < 2; ++ni) {
            const int col = col0 + wn * 32 + ni * 16 + ln;
#pragma unroll
            for (int r = 0; r < 8; ++r) {
                const int row = row0 + wm * 64 + mi * 16 + half * 8 + r;
                if (OUT_BF16)
                    ((__bf16*)Cout)[(size_t)row * N + col] = (__bf16)acc[mi][ni][r];
                else
                    ((float*)Cout)[(size_t)row * N + col] = acc[mi][ni][r];
            }
        }
}

// ---------------------------------------------------------------------------
// Llama3-scaled RoPE, applied in place to Q (cols 0..4095) and K (4096..5119).
// Head-slot 0..31 = Q heads, 32..39 = K heads (K is contiguous after Q).
// ---------------------------------------------------------------------------
__global__ void rope_kernel(__bf16* __restrict__ qkv, const int* __restrict__ pos) {
    const long total = (long)MROWS * 40 * 64;
    const long i = (long)blockIdx.x * blockDim.x + threadIdx.x;
    if (i >= total) return;
    const int d   = (int)(i % 64);
    const long t  = i / 64;
    const int hs  = (int)(t % 40);
    const int row = (int)(t / 40);

    // llama3 inv-freq scaling (OLD_CTX=8192, factor 8, low=1, high=4)
    const float fr  = (float)(2 * d) * (1.0f / 128.0f);
    const float inv = __expf(-fr * 13.122363377404328f);      // 500000^(-2d/128)
    const float wavelen = 6.283185307179586f / inv;
    float inv_used;
    if (wavelen > 8192.0f)       inv_used = inv * 0.125f;
    else if (wavelen < 2048.0f)  inv_used = inv;
    else {
        const float smooth = (8192.0f / wavelen - 1.0f) * (1.0f / 3.0f);
        inv_used = (1.0f - smooth) * inv * 0.125f + smooth * inv;
    }
    const float f = (float)pos[row] * inv_used;
    float s, c;
    __sincosf(f, &s, &c);

    __bf16* p = qkv + (size_t)row * QKVD + hs * HEAD_DIM + d;
    const float x1 = (float)p[0];
    const float x2 = (float)p[64];
    p[0]  = (__bf16)(x1 * c - x2 * s);
    p[64] = (__bf16)(x2 * c + x1 * s);
}

// ---------------------------------------------------------------------------
// Flash attention (GQA, causal). One block per (b, kv-head, 16-query block);
// the 4 waves are the 4 query heads of the GQA group and SHARE the K/V LDS
// tiles. S = Q*K^T via 4 chained bf16 WMMAs per 16x16 tile; online softmax
// with 16-lane shfl_xor row reductions; P relayouted C-frag -> A-frag via
// LDS; P*V via LDS-transposed V (fragments = two ds_load_b128 each).
// ---------------------------------------------------------------------------
__global__ void __launch_bounds__(128)
flash_attn(const __bf16* __restrict__ qkv, __bf16* __restrict__ out) {
    __shared__ __attribute__((aligned(16))) __bf16 Ks[32][136];      // keys x d
    __shared__ __attribute__((aligned(16))) __bf16 VsT[128][40];     // d x keys
    __shared__ __attribute__((aligned(16))) __bf16 Ps[4][16][40];    // per-wave P

    const int bid  = blockIdx.x;
    const int qblk = bid & 127;            // 16-row query block
    const int kvh  = (bid >> 7) & 7;
    const int b    = bid >> 10;

    const int lane = threadIdx.x & 31;
    const int wave = threadIdx.x >> 5;
    const int half = lane >> 4;
    const int ln   = lane & 15;
    const int qh   = kvh * 4 + wave;

    // Q fragments straight from global (rows are 16B-aligned chunks)
    const __bf16* qbase = qkv + (size_t)(b * SEQ + qblk * 16) * QKVD + qh * HEAD_DIM;
    bf16x16 qf[4];
#pragma unroll
    for (int c = 0; c < 4; ++c)
        qf[c] = make_frag(qbase + (size_t)ln * QKVD + c * 32 + half * 8,
                          qbase + (size_t)ln * QKVD + c * 32 + 16 + half * 8);

    float m[8], l[8];
    f32x8 o[8] = {};
#pragma unroll
    for (int r = 0; r < 8; ++r) { m[r] = -1e30f; l[r] = 0.0f; }

    const int tkey = threadIdx.x >> 2;          // 0..31
    const int toff = (threadIdx.x & 3) * 32;    // 0,32,64,96
    const int jmax = (qblk * 16 + 15) >> 5;     // inclusive

    for (int j = 0; j <= jmax; ++j) {
        // ---- stage K (row-major) and V (transposed) into LDS --------------
        const size_t krow = (size_t)(b * SEQ + j * 32 + tkey) * QKVD + kvh * HEAD_DIM;
        const __bf16* ksrc = qkv + krow + KOFF + toff;
        const __bf16* vsrc = qkv + krow + VOFF + toff;
#pragma unroll
        for (int e = 0; e < 4; ++e)
            *(bf16x8*)&Ks[tkey][toff + e * 8] = *(const bf16x8*)(ksrc + e * 8);
        bf16x8 vv[4];
#pragma unroll
        for (int e = 0; e < 4; ++e) vv[e] = *(const bf16x8*)(vsrc + e * 8);
#pragma unroll
        for (int e = 0; e < 4; ++e)
#pragma unroll
            for (int i = 0; i < 8; ++i)
                VsT[toff + e * 8 + i][tkey] = vv[e][i];
        __syncthreads();

        // ---- S = Q*K^T (two 16x16 key tiles, K-dim=128 = 4 WMMAs each) ----
        f32x8 s[2] = {};
#pragma unroll
        for (int t = 0; t < 2; ++t)
#pragma unroll
            for (int c = 0; c < 4; ++c) {
                const int kr = t * 16 + ln;
                bf16x16 kf = make_frag(&Ks[kr][c * 32 + half * 8],
                                       &Ks[kr][c * 32 + 16 + half * 8]);
                s[t] = __builtin_amdgcn_wmma_f32_16x16x32_bf16(
                    false, qf[c], false, kf, (short)0, s[t], false, false);
            }

        // ---- scale + causal mask ------------------------------------------
#pragma unroll
        for (int t = 0; t < 2; ++t)
#pragma unroll
            for (int r = 0; r < 8; ++r) {
                float v = s[t][r] * 0.08838834764831845f;     // 1/sqrt(128)
                const int key = j * 32 + t * 16 + ln;
                const int qr  = qblk * 16 + half * 8 + r;
                s[t][r] = (key > qr) ? -1e30f : v;
            }

        // ---- online softmax (row reductions across the 16-lane half) ------
        float alpha[8];
#pragma unroll
        for (int r = 0; r < 8; ++r) {
            float mx = fmaxf(s[0][r], s[1][r]);
#pragma unroll
            for (int off = 1; off < 16; off <<= 1)
                mx = fmaxf(mx, __shfl_xor(mx, off, 32));
            const float mn = fmaxf(m[r], mx);
            alpha[r] = __expf(m[r] - mn);
            m[r] = mn;
        }
#pragma unroll
        for (int t = 0; t < 2; ++t)
#pragma unroll
            for (int r = 0; r < 8; ++r)
                s[t][r] = __expf(s[t][r] - m[r]);
#pragma unroll
        for (int r = 0; r < 8; ++r) {
            float ps = s[0][r] + s[1][r];
#pragma unroll
            for (int off = 1; off < 16; off <<= 1)
                ps += __shfl_xor(ps, off, 32);
            l[r] = l[r] * alpha[r] + ps;
        }

        // ---- P: C-layout f32 -> A-layout bf16 via per-wave LDS ------------
#pragma unroll
        for (int t = 0; t < 2; ++t)
#pragma unroll
            for (int r = 0; r < 8; ++r)
                Ps[wave][half * 8 + r][t * 16 + ln] = (__bf16)s[t][r];
        asm volatile("s_wait_dscnt 0x0" ::: "memory");  // wave-local LDS RAW fence
        bf16x16 pf = make_frag(&Ps[wave][ln][half * 8],
                               &Ps[wave][ln][16 + half * 8]);

        // ---- O = O*alpha + P*V (8 head-dim tiles) --------------------------
#pragma unroll
        for (int dt = 0; dt < 8; ++dt) {
#pragma unroll
            for (int r = 0; r < 8; ++r) o[dt][r] *= alpha[r];
            const int dr = dt * 16 + ln;
            bf16x16 vf = make_frag(&VsT[dr][half * 8], &VsT[dr][16 + half * 8]);
            o[dt] = __builtin_amdgcn_wmma_f32_16x16x32_bf16(
                false, pf, false, vf, (short)0, o[dt], false, false);
        }
        __syncthreads();   // before next K/V staging
    }

    // ---- epilogue: normalize, store bf16 for the WO GEMM -------------------
#pragma unroll
    for (int dt = 0; dt < 8; ++dt)
#pragma unroll
        for (int r = 0; r < 8; ++r) {
            const int row = b * SEQ + qblk * 16 + half * 8 + r;
            const int col = qh * HEAD_DIM + dt * 16 + ln;
            out[(size_t)row * (NQH * HEAD_DIM) + col] = (__bf16)(o[dt][r] / l[r]);
        }
}

// ---------------------------------------------------------------------------
// Launch: cast/transpose -> QKV GEMM -> RoPE -> flash attention -> WO GEMM
// Workspace layout (bytes): xb 32M | wqkvT 48M | woT 32M | qkv 48M | ao 32M
// ---------------------------------------------------------------------------
extern "C" void kernel_launch(void* const* d_in, const int* in_sizes, int n_in,
                              void* d_out, int out_size, void* d_ws, size_t ws_size,
                              hipStream_t stream) {
    (void)in_sizes; (void)n_in; (void)out_size; (void)ws_size;
    const float* x    = (const float*)d_in[0];
    const int*   pos  = (const int*)d_in[1];
    const float* wqkv = (const float*)d_in[2];
    const float* wo   = (const float*)d_in[3];
    float* out = (float*)d_out;

    __bf16* xb    = (__bf16*)d_ws;
    __bf16* wqkvT = xb    + (size_t)MROWS * HDIM;
    __bf16* woT   = wqkvT + (size_t)QKVD * HDIM;
    __bf16* qkv   = woT   + (size_t)HDIM * HDIM;
    __bf16* ao    = qkv   + (size_t)MROWS * QKVD;

    cast_bf16<<<4096, 256, 0, stream>>>(x, xb, (long)MROWS * HDIM);
    transpose_bf16<<<8192, 256, 0, stream>>>(wqkv, wqkvT, HDIM, QKVD);
    transpose_bf16<<<8192, 256, 0, stream>>>(wo, woT, HDIM, HDIM);

    gemm_bf16_wmma<true><<<dim3(QKVD / 128, MROWS / 128), 256, 0, stream>>>(
        xb, wqkvT, qkv, MROWS, QKVD, HDIM);

    const long rope_n = (long)MROWS * 40 * 64;
    rope_kernel<<<(int)((rope_n + 255) / 256), 256, 0, stream>>>(qkv, pos);

    flash_attn<<<BATCH * NKV * (SEQ / 16), 128, 0, stream>>>(qkv, ao);

    gemm_bf16_wmma<false><<<dim3(HDIM / 128, MROWS / 128), 256, 0, stream>>>(
        ao, woT, out, MROWS, HDIM, HDIM);
}
